// GraphAttentionNetwork_51127290691641
// MI455X (gfx1250) — matrix operand order repaired and checked by reading, not compiled, not run on hardware
//
#include <hip/hip_runtime.h>
#include <hip/hip_bf16.h>

// ---------------- problem constants (from reference) ----------------
#define NNODES 50000
#define NEDGES 800000
#define EALL   (NEDGES + NNODES)   // self-loops appended
#define INCH   128
#define HID    32
#define HEADS  4
#define HC     (HID * HEADS)       // 128
#define LATENT 64
#define NEGSLOPE 0.2f

typedef __attribute__((ext_vector_type(16))) _Float16 v16h;
typedef __attribute__((ext_vector_type(8)))  float    v8f;

// ---------------- A-fragment loader (contiguous k order, same as B loads) ----------------
// A: 16x32 f16 (MxK).  lane m = lane&15.
//   lanes 0-15 : halves 0..15 -> K=k0+0..15
//   lanes 16-31: halves 0..15 -> K=k0+16..31
__device__ inline v16h load_a_frag(const _Float16* __restrict__ A, int K,
                                   int row0, int k0, int lane) {
    int m  = row0 + (lane & 15);
    int kb = k0 + ((lane >> 4) << 4);
    const _Float16* p = A + (size_t)m * K + kb;
    v16h a;
#pragma unroll
    for (int j = 0; j < 8; ++j) { a[j] = p[j]; a[j + 8] = p[8 + j]; }
    return a;
}

// ---------------- row-block WMMA GEMM ----------------
// out[M x N] = act(A[M x K] * B[K x N] + bias), B given transposed (Bt[n*K+k]).
// Each wave owns a 16-row x N-col row block. Per k-step: load A fragment once,
// load ALL NT B fragments (distinct live ranges -> single ds clause + one wait),
// then issue NT back-to-back WMMAs. Weights staged in LDS once per block.
template <int NT, int KK, int ACT, int OUTF16>
__global__ void wmma_gemm_row_kernel(const _Float16* __restrict__ A,
                                     const _Float16* __restrict__ Bt,
                                     const float* __restrict__ bias,
                                     void* __restrict__ outp,
                                     int Mtiles) {
    constexpr int N = NT * 16;
    __shared__ _Float16 Bs[N * KK];          // <= 32 KB

    // cooperative copy of the (tiny) weight matrix into LDS
    {
        constexpr int NVEC = (N * KK) / 8;   // 8 halves per uint4
        const uint4* src = (const uint4*)Bt;
        uint4* dst = (uint4*)Bs;
        for (int i = threadIdx.x; i < NVEC; i += blockDim.x) dst[i] = src[i];
    }
    __syncthreads();

    int lane = threadIdx.x & 31;
    int wave = threadIdx.x >> 5;
    int rb   = blockIdx.x * (blockDim.x >> 5) + wave;
    if (rb >= Mtiles) return;                // uniform per wave -> EXEC all 1s inside
    int row0 = rb << 4;

    v8f c[NT] = {};

    // B fragment layout: 32x16 f16 (KxN). lane n = lane&15,
    // halves j -> K = k0 + 16*(lane>=16) + j  (contiguous in Bt[n*KK + k])
    int bn = lane & 15;
    int bo = (lane >> 4) << 4;

#pragma unroll
    for (int k0 = 0; k0 < KK; k0 += 32) {
        v16h a = load_a_frag(A, KK, row0, k0, lane);

        v16h b[NT];
#pragma unroll
        for (int t = 0; t < NT; ++t) {
            const _Float16* bp = &Bs[(t * 16 + bn) * KK + k0 + bo];
#pragma unroll
            for (int j = 0; j < 16; ++j) b[t][j] = bp[j];
        }
#pragma unroll
        for (int t = 0; t < NT; ++t) {
            c[t] = __builtin_amdgcn_wmma_f32_16x16x32_f16(false, a, false, b[t],
                                                          (short)0, c[t], false, false);
        }
    }

    // C/D layout: VGPR r -> row = row0 + r + 8*(lane>=16), col = n0 + (lane&15)
    int col16 = lane & 15;
    int rbase = row0 + ((lane >> 4) << 3);
#pragma unroll
    for (int t = 0; t < NT; ++t) {
        int col  = t * 16 + col16;
        float bv = bias ? bias[col] : 0.f;
#pragma unroll
        for (int r = 0; r < 8; ++r) {
            float v = c[t][r] + bv;
            if (ACT == 1) v = (v > 0.f) ? v : (__expf(v) - 1.f);   // ELU
            size_t idx = (size_t)(rbase + r) * N + col;
            if (OUTF16) ((_Float16*)outp)[idx] = (_Float16)v;
            else        ((float*)outp)[idx]    = v;
        }
    }
}

// ---------------- conversion / prep kernels ----------------
__global__ void cvt_f16_kernel(const float* __restrict__ src, _Float16* __restrict__ dst, int n) {
    int i = blockIdx.x * blockDim.x + threadIdx.x;
    if (i < n) dst[i] = (_Float16)src[i];
}

// W[K x N] f32 row-major -> Wt[N x K] f16 (transposed)
__global__ void cvt_f16_t_kernel(const float* __restrict__ W, _Float16* __restrict__ Wt, int K, int N) {
    int i = blockIdx.x * blockDim.x + threadIdx.x;
    if (i >= K * N) return;
    int k = i / N, n = i % N;
    Wt[(size_t)n * K + k] = (_Float16)W[i];
}

__global__ void init_kernel(float* __restrict__ agg, float* __restrict__ denom,
                            float* __restrict__ emax, int nAgg, int nH) {
    int i = blockIdx.x * blockDim.x + threadIdx.x;
    if (i < nAgg) agg[i] = 0.f;
    if (i < nH) { denom[i] = 0.f; emax[i] = -__builtin_inff(); }
}

// a_src[n,h] = dot(h[n,h,:], att_src[h,:]) ; same for dst. thread per (node,head)
__global__ void attn_coeff_kernel(const float* __restrict__ h,
                                  const float* __restrict__ att_s,
                                  const float* __restrict__ att_d,
                                  float* __restrict__ a_s,
                                  float* __restrict__ a_d) {
    int i = blockIdx.x * blockDim.x + threadIdx.x;
    if (i >= NNODES * HEADS) return;
    int n = i >> 2, hd = i & 3;
    const float* hp = h + (size_t)n * HC + hd * HID;
    const float* as = att_s + hd * HID;
    const float* ad = att_d + hd * HID;
    float ss = 0.f, sd = 0.f;
#pragma unroll
    for (int c = 0; c < HID; ++c) { ss += hp[c] * as[c]; sd += hp[c] * ad[c]; }
    a_s[i] = ss; a_d[i] = sd;
}

__device__ inline void atomicMaxF(float* addr, float v) {
    if (v >= 0.f) atomicMax((int*)addr, __float_as_int(v));
    else          atomicMin((unsigned int*)addr, __float_as_uint(v));
}

__device__ inline void edge_sd(const long long* __restrict__ ei, int e, int& s, int& d) {
    if (e < NEDGES) { s = (int)ei[e]; d = (int)ei[NEDGES + e]; }
    else            { s = e - NEDGES; d = s; }
}

// pass 1: segment max of leaky_relu(a_src[s]+a_dst[d])
__global__ void edge_max_kernel(const long long* __restrict__ ei,
                                const float* __restrict__ a_s,
                                const float* __restrict__ a_d,
                                float* __restrict__ emax) {
    int e = blockIdx.x * blockDim.x + threadIdx.x;
    if (e >= EALL) return;
    int s, d; edge_sd(ei, e, s, d);
#pragma unroll
    for (int hd = 0; hd < HEADS; ++hd) {
        float t = a_s[s * HEADS + hd] + a_d[d * HEADS + hd];
        t = (t > 0.f) ? t : NEGSLOPE * t;
        atomicMaxF(&emax[d * HEADS + hd], t);
    }
}

// pass 2: e_exp = exp(e - max[d]); denom[d] += e_exp
__global__ void edge_exp_kernel(const long long* __restrict__ ei,
                                const float* __restrict__ a_s,
                                const float* __restrict__ a_d,
                                const float* __restrict__ emax,
                                float* __restrict__ e_exp,
                                float* __restrict__ denom) {
    int e = blockIdx.x * blockDim.x + threadIdx.x;
    if (e >= EALL) return;
    int s, d; edge_sd(ei, e, s, d);
#pragma unroll
    for (int hd = 0; hd < HEADS; ++hd) {
        float t = a_s[s * HEADS + hd] + a_d[d * HEADS + hd];
        t = (t > 0.f) ? t : NEGSLOPE * t;
        float w = __expf(t - emax[d * HEADS + hd]);
        e_exp[(size_t)e * HEADS + hd] = w;
        atomicAdd(&denom[d * HEADS + hd], w);
    }
}

// pass 3: agg[d,c] += h[s,c] * alpha[e, head(c)] ; 32 lanes per edge, 4 channels/lane
__global__ void edge_msg_kernel(const long long* __restrict__ ei,
                                const float* __restrict__ h,
                                const float* __restrict__ e_exp,
                                const float* __restrict__ denom,
                                float* __restrict__ agg) {
    int t = blockIdx.x * blockDim.x + threadIdx.x;
    int e = t >> 5, lane = t & 31;
    if (e >= EALL) return;
    int s, d; edge_sd(ei, e, s, d);
    int c0 = lane * 4;
    int hd = c0 >> 5;   // 32 channels per head
    float alpha = e_exp[(size_t)e * HEADS + hd] /
                  (denom[d * HEADS + hd] + 1e-16f);
    const float4 hv = *(const float4*)(h + (size_t)s * HC + c0);
    float* ap = agg + (size_t)d * HC + c0;
    atomicAdd(ap + 0, hv.x * alpha);
    atomicAdd(ap + 1, hv.y * alpha);
    atomicAdd(ap + 2, hv.z * alpha);
    atomicAdd(ap + 3, hv.w * alpha);
}

// x1 = f16(elu(agg + b_gat))
__global__ void act1_kernel(const float* __restrict__ agg, const float* __restrict__ b_gat,
                            _Float16* __restrict__ x1, int n) {
    int i = blockIdx.x * blockDim.x + threadIdx.x;
    if (i >= n) return;
    float v = agg[i] + b_gat[i & (HC - 1)];
    v = (v > 0.f) ? v : (__expf(v) - 1.f);
    x1[i] = (_Float16)v;
}

// ---------------- host launch ----------------
extern "C" void kernel_launch(void* const* d_in, const int* in_sizes, int n_in,
                              void* d_out, int out_size, void* d_ws, size_t ws_size,
                              hipStream_t stream) {
    const float*     x        = (const float*)d_in[0];
    const long long* ei       = (const long long*)d_in[1];   // int64 edge_index [2,E]
    const float*     W_gat    = (const float*)d_in[2];
    const float*     b_gat    = (const float*)d_in[3];
    const float*     att_src  = (const float*)d_in[4];
    const float*     att_dst  = (const float*)d_in[5];
    const float*     W_enc    = (const float*)d_in[6];
    const float*     b_enc    = (const float*)d_in[7];
    const float*     W_dec    = (const float*)d_in[8];
    const float*     b_dec    = (const float*)d_in[9];
    const float*     W_out    = (const float*)d_in[10];
    const float*     b_out    = (const float*)d_in[11];
    float*           out      = (float*)d_out;

    char* wsb = (char*)d_ws;
    size_t off = 0;
    auto alloc = [&](size_t bytes) -> void* {
        void* p = wsb + off;
        off = (off + bytes + 255) & ~(size_t)255;
        return p;
    };
    _Float16* xf16   = (_Float16*)alloc((size_t)NNODES * INCH * 2);
    _Float16* wgat_t = (_Float16*)alloc((size_t)INCH * HC * 2);
    _Float16* wenc_t = (_Float16*)alloc((size_t)HC * LATENT * 2);
    _Float16* wdec_t = (_Float16*)alloc((size_t)LATENT * HC * 2);
    _Float16* wout_t = (_Float16*)alloc((size_t)HC * INCH * 2);
    float*    h      = (float*)alloc((size_t)NNODES * HC * 4);
    float*    a_s    = (float*)alloc((size_t)NNODES * HEADS * 4);
    float*    a_d    = (float*)alloc((size_t)NNODES * HEADS * 4);
    float*    emax   = (float*)alloc((size_t)NNODES * HEADS * 4);
    float*    denom  = (float*)alloc((size_t)NNODES * HEADS * 4);
    float*    e_exp  = (float*)alloc((size_t)EALL * HEADS * 4);
    float*    agg    = (float*)alloc((size_t)NNODES * HC * 4);
    _Float16* x1f16  = (_Float16*)alloc((size_t)NNODES * HC * 2);
    _Float16* encf16 = (_Float16*)alloc((size_t)NNODES * LATENT * 2);
    _Float16* x2f16  = (_Float16*)alloc((size_t)NNODES * HC * 2);
    (void)ws_size; (void)n_in; (void)in_sizes; (void)out_size;

    const int T = 256;
    const int WPB = T / 32;                    // waves per block
    const int MT = NNODES / 16;                // 3125 row tiles
    auto blks = [](long long n, int t) { return (int)((n + t - 1) / t); };
    int gemmBlocks = (MT + WPB - 1) / WPB;

    // prep: f16 conversions
    cvt_f16_kernel<<<blks((long long)NNODES * INCH, T), T, 0, stream>>>(x, xf16, NNODES * INCH);
    cvt_f16_t_kernel<<<blks(INCH * HC, T), T, 0, stream>>>(W_gat, wgat_t, INCH, HC);
    cvt_f16_t_kernel<<<blks(HC * LATENT, T), T, 0, stream>>>(W_enc, wenc_t, HC, LATENT);
    cvt_f16_t_kernel<<<blks(LATENT * HC, T), T, 0, stream>>>(W_dec, wdec_t, LATENT, HC);
    cvt_f16_t_kernel<<<blks(HC * INCH, T), T, 0, stream>>>(W_out, wout_t, HC, INCH);
    init_kernel<<<blks((long long)NNODES * HC, T), T, 0, stream>>>(agg, denom, emax,
                                                                   NNODES * HC, NNODES * HEADS);

    // h = x @ W_gat   (M=50000, N=128, K=128) -> f32
    wmma_gemm_row_kernel<8, 128, 0, 0><<<gemmBlocks, T, 0, stream>>>(
        xf16, wgat_t, nullptr, h, MT);

    attn_coeff_kernel<<<blks((long long)NNODES * HEADS, T), T, 0, stream>>>(h, att_src, att_dst, a_s, a_d);

    edge_max_kernel<<<blks(EALL, T), T, 0, stream>>>(ei, a_s, a_d, emax);
    edge_exp_kernel<<<blks(EALL, T), T, 0, stream>>>(ei, a_s, a_d, emax, e_exp, denom);
    edge_msg_kernel<<<blks((long long)EALL * 32, T), T, 0, stream>>>(ei, h, e_exp, denom, agg);

    act1_kernel<<<blks((long long)NNODES * HC, T), T, 0, stream>>>(agg, b_gat, x1f16, NNODES * HC);

    // encoded = x1 @ W_enc + b_enc  (N=64, K=128) -> f16
    wmma_gemm_row_kernel<4, 128, 0, 1><<<gemmBlocks, T, 0, stream>>>(
        x1f16, wenc_t, b_enc, encf16, MT);

    // x2 = elu(encoded @ W_dec + b_dec)  (N=128, K=64) -> f16
    wmma_gemm_row_kernel<8, 64, 1, 1><<<gemmBlocks, T, 0, stream>>>(
        encf16, wdec_t, b_dec, x2f16, MT);

    // out = x2 @ W_out + b_out  (N=128, K=128) -> f32 d_out
    wmma_gemm_row_kernel<8, 128, 0, 0><<<gemmBlocks, T, 0, stream>>>(
        x2f16, wout_t, b_out, out, MT);
}
